// SelfAttention_67637144978282
// MI455X (gfx1250) — compile-verified, matrix operand order checked by reading, not run on hardware
//
#include <hip/hip_runtime.h>
#include <stdint.h>
#include <stddef.h>

// ---------------------------------------------------------------------------
// Self-attention (B=128, S=128, D=1024) for gfx1250 using bf16 WMMA with
// async global->LDS staging (ASYNCcnt pipeline) for every shared operand.
// ---------------------------------------------------------------------------

typedef __bf16 bf16;
typedef __attribute__((ext_vector_type(16))) __bf16 v16bf;
typedef __attribute__((ext_vector_type(8)))  __bf16 v8bf;
typedef __attribute__((ext_vector_type(4)))  __bf16 v4bf;
typedef __attribute__((ext_vector_type(8)))  float  v8f;

#define D_EMB  1024
#define SEQ    128
#define NBATCH 128
#define ROWS   (NBATCH * SEQ)    // 16384
#define SLD    132               // padded LDS stride for fp32 scores
#define TLD    40                // padded LDS row stride (bf16) for 32-wide tiles
#define VLD    136               // padded LDS row stride (bf16) for 128-wide V tiles

// --------------------------- helpers ---------------------------------------

__device__ __forceinline__ v8f vzero8f() {
    v8f z;
#pragma unroll
    for (int i = 0; i < 8; ++i) z[i] = 0.0f;
    return z;
}

// Load a 16x32 bf16 fragment from a row-major matrix (stride `ld` elements).
// CDNA5 ISA "16-bit A-Matrix 16x32" layout:
//   lane L holds row/col r = L&15; k-base = (L>>4)*8
//   elements 0..7  -> K = kbase + 0..7    (contiguous 16B)
//   elements 8..15 -> K = kbase + 16..23  (contiguous 16B)
// Same layout serves B when B[k][n] = M[n][k] for row-major M (X * M^T).
__device__ __forceinline__ v16bf load_frag16x32(const bf16* __restrict__ base,
                                                int ld, int lane) {
    const int r  = lane & 15;
    const int kb = (lane >> 4) << 3;            // 0 or 8
    const bf16* p = base + (size_t)r * ld + kb;
    const v8bf lo = *reinterpret_cast<const v8bf*>(p);
    const v8bf hi = *reinterpret_cast<const v8bf*>(p + 16);
    v16bf f;
#pragma unroll
    for (int i = 0; i < 8; ++i) { f[i] = lo[i]; f[i + 8] = hi[i]; }
    return f;
}

__device__ __forceinline__ v8f wmma_bf16(v16bf a, v16bf b, v8f c) {
    return __builtin_amdgcn_wmma_f32_16x16x32_bf16(false, a, false, b,
                                                   (short)0, c, false, false);
}

// Async global -> LDS copy, 16 bytes per lane (ASYNCcnt-tracked).
__device__ __forceinline__ void async_load_b128(uint32_t lds_byte_off,
                                                const bf16* g) {
    asm volatile("global_load_async_to_lds_b128 %0, %1, off"
                 :: "v"(lds_byte_off), "v"(g)
                 : "memory");
}

__device__ __forceinline__ void wait_asynccnt0() {
    asm volatile("s_wait_asynccnt 0x0" ::: "memory");
}

__device__ __forceinline__ uint32_t lds_off_u32(const void* p) {
    // On amdgcn the generic address of an LDS object carries the LDS byte
    // offset in its low 32 bits (aperture in the high bits).
    return (uint32_t)(uintptr_t)p;
}

// --------------------------- conversion ------------------------------------

__global__ void cvt_f32_to_bf16_x4(const float* __restrict__ src,
                                   bf16* __restrict__ dst, int n4) {
    int i = blockIdx.x * blockDim.x + threadIdx.x;
    if (i < n4) {
        const float4 v = reinterpret_cast<const float4*>(src)[i];
        v4bf o;
        o[0] = (bf16)v.x; o[1] = (bf16)v.y; o[2] = (bf16)v.z; o[3] = (bf16)v.w;
        reinterpret_cast<v4bf*>(dst)[i] = o;
    }
}

// --------------------------- QKV projections --------------------------------
// 8 waves per block; block computes a 128x64 output tile. The 64x32 weight
// tile per K-step is shared by all waves: async-copied to LDS, double
// buffered, copy of step k+1 overlapped with WMMAs of step k.
// grid = (ROWS/128, D_EMB/64, 3)  z: 0=Q (bias+1/sqrt(d)), 1=K, 2=V^T
__global__ void qkv_gemm(const bf16* __restrict__ xbf,
                         const bf16* __restrict__ wq,
                         const bf16* __restrict__ wk,
                         const bf16* __restrict__ wv,
                         const float* __restrict__ bq,
                         const float* __restrict__ bk,
                         const float* __restrict__ bv,
                         bf16* __restrict__ qout,
                         bf16* __restrict__ kout,
                         bf16* __restrict__ vtout) {
    __shared__ bf16 wtile[2][64 * TLD];

    const int z = blockIdx.z;
    const bf16*  W    = (z == 0) ? wq : (z == 1) ? wk : wv;
    const float* bias = (z == 0) ? bq : (z == 1) ? bk : bv;

    const int tid  = threadIdx.x;
    const int lane = tid & 31;
    const int w    = tid >> 5;
    const int r0   = blockIdx.x * 128 + w * 16;
    const int c0   = blockIdx.y * 64;

    const int trow = tid >> 2;   // 0..63 : row of the 64x32 weight tile
    const int tch  = tid & 3;    // 16B chunk within the 64B row
    const uint32_t wt_base = lds_off_u32(&wtile[0][0]);

    v8f acc[4];
#pragma unroll
    for (int t = 0; t < 4; ++t) acc[t] = vzero8f();

    const bf16* arow = xbf + (size_t)r0 * D_EMB;

    // prologue: stage K-step 0
    async_load_b128(wt_base + (uint32_t)(trow * (TLD * 2) + tch * 16),
                    W + (size_t)(c0 + trow) * D_EMB + tch * 8);
    wait_asynccnt0();
    __syncthreads();

    v16bf a_cur = load_frag16x32(arow, D_EMB, lane);

    for (int kt = 0; kt < 32; ++kt) {
        const int  k0   = kt * 32;
        const bool more = (kt + 1) < 32;

        if (more) {   // overlap next tile copy with this step's math
            const uint32_t buf = (uint32_t)((kt + 1) & 1) * (64 * TLD * 2);
            async_load_b128(wt_base + buf + (uint32_t)(trow * (TLD * 2) + tch * 16),
                            W + (size_t)(c0 + trow) * D_EMB + k0 + 32 + tch * 8);
        }
        v16bf a_nxt = a_cur;
        if (more) {
            __builtin_prefetch(arow + k0 + 64, 0, 1);
            a_nxt = load_frag16x32(arow + k0 + 32, D_EMB, lane);
        }

        const bf16* wt = &wtile[kt & 1][0];
#pragma unroll
        for (int t = 0; t < 4; ++t) {
            const v16bf b = load_frag16x32(wt + t * 16 * TLD, TLD, lane);
            acc[t] = wmma_bf16(a_cur, b, acc[t]);
        }
        a_cur = a_nxt;

        if (more) {
            wait_asynccnt0();    // next tile has landed (copy overlapped)
            __syncthreads();
        }
    }

    // Epilogue: C layout => lane holds column n = lane&15, rows mb..mb+7.
    const int n  = lane & 15;
    const int mb = (lane >> 4) * 8;
#pragma unroll
    for (int t = 0; t < 4; ++t) {
        const int   col = c0 + t * 16 + n;
        const float bb  = bias[col];
#pragma unroll
        for (int j = 0; j < 8; ++j) {
            const int   row = r0 + mb + j;
            const float r   = acc[t][j] + bb;
            if (z == 0) {
                qout[(size_t)row * D_EMB + col] = (bf16)(r * 0.03125f);
            } else if (z == 1) {
                kout[(size_t)row * D_EMB + col] = (bf16)r;
            } else {
                const int batch = row >> 7;
                const int s     = row & 127;
                vtout[(size_t)batch * D_EMB * SEQ + (size_t)col * SEQ + s] = (bf16)r;
            }
        }
    }
}

// --------------------------- fused attention --------------------------------
// One block (8 waves) per batch.
//   phase 1: scores = Q*K^T with the 128x32 K-tile async-staged in LDS
//   phase 2: row softmax in LDS (V-tile 0 DMA overlapped with it)
//   phase 3: attended = P*V with 32x128 V^T tiles async-staged in LDS
__global__ void attn_kernel(const bf16* __restrict__ q,
                            const bf16* __restrict__ kmat,
                            const bf16* __restrict__ vt,
                            bf16* __restrict__ att) {
    extern __shared__ char smem[];
    float* sc    = reinterpret_cast<float*>(smem);                        // 128 x SLD f32
    bf16*  P     = reinterpret_cast<bf16*>(smem + SEQ * SLD * sizeof(float)); // 128x128
    bf16*  ktile = reinterpret_cast<bf16*>(smem + SEQ * SLD * sizeof(float)
                                                + SEQ * SEQ * sizeof(bf16)); // 2x128xTLD
    bf16*  vtile = ktile;   // reused after phase 1: 2 x 32 x VLD (17408B <= 20480B)

    const int b    = blockIdx.x;
    const int tid  = threadIdx.x;
    const int lane = tid & 31;
    const int w    = tid >> 5;
    const int s0   = w * 16;

    const bf16* Qb  = q    + (size_t)b * SEQ * D_EMB;
    const bf16* Kb  = kmat + (size_t)b * SEQ * D_EMB;
    const bf16* Vtb = vt   + (size_t)b * D_EMB * SEQ;

    const uint32_t kt_base = lds_off_u32(ktile);

    // ---- phase 1: scores = Q * K^T  (Q pre-scaled by 1/sqrt(D)) ----
    v8f acc[8];
#pragma unroll
    for (int t = 0; t < 8; ++t) acc[t] = vzero8f();

    const bf16* qrow = Qb + (size_t)s0 * D_EMB;

    // stage K-tile 0 (128 rows x 32 cols = 512 b128 chunks, 2 per thread)
#pragma unroll
    for (int j = 0; j < 2; ++j) {
        const int chunk = tid + j * 256;
        const int krow  = chunk >> 2;
        const int kch   = chunk & 3;
        async_load_b128(kt_base + (uint32_t)(krow * (TLD * 2) + kch * 16),
                        Kb + (size_t)krow * D_EMB + kch * 8);
    }
    wait_asynccnt0();
    __syncthreads();

    v16bf a_cur = load_frag16x32(qrow, D_EMB, lane);

    for (int kt = 0; kt < 32; ++kt) {
        const int  k0   = kt * 32;
        const bool more = (kt + 1) < 32;

        if (more) {
            const uint32_t buf = (uint32_t)((kt + 1) & 1) * (SEQ * TLD * 2);
#pragma unroll
            for (int j = 0; j < 2; ++j) {
                const int chunk = tid + j * 256;
                const int krow  = chunk >> 2;
                const int kch   = chunk & 3;
                async_load_b128(kt_base + buf + (uint32_t)(krow * (TLD * 2) + kch * 16),
                                Kb + (size_t)krow * D_EMB + k0 + 32 + kch * 8);
            }
        }
        v16bf a_nxt = a_cur;
        if (more) a_nxt = load_frag16x32(qrow + k0 + 32, D_EMB, lane);

        const bf16* kl = ktile + (kt & 1) * (SEQ * TLD);
#pragma unroll
        for (int t = 0; t < 8; ++t) {
            const v16bf bf = load_frag16x32(kl + t * 16 * TLD, TLD, lane);
            acc[t] = wmma_bf16(a_cur, bf, acc[t]);
        }
        a_cur = a_nxt;

        if (more) {
            wait_asynccnt0();
            __syncthreads();
        }
    }
    const int n  = lane & 15;
    const int mb = (lane >> 4) * 8;
#pragma unroll
    for (int t = 0; t < 8; ++t)
#pragma unroll
        for (int j = 0; j < 8; ++j)
            sc[(s0 + mb + j) * SLD + t * 16 + n] = acc[t][j];
    __syncthreads();

    // ---- phase 2: overlap V-tile 0 DMA with the softmax ----
    // V tile for output cols [0,32): 32 rows x 128 cols = 512 chunks, 2/thread
#pragma unroll
    for (int j = 0; j < 2; ++j) {
        const int chunk = tid + j * 256;
        const int vrow  = chunk >> 4;           // 0..31
        const int vch   = chunk & 15;           // 16B chunk in the 256B row
        async_load_b128(kt_base + (uint32_t)(vrow * (VLD * 2) + vch * 16),
                        Vtb + (size_t)vrow * SEQ + vch * 8);
    }

    if (tid < SEQ) {
        float* row = sc + tid * SLD;
        float  m = row[0];
        for (int i = 1; i < SEQ; ++i) m = fmaxf(m, row[i]);
        float s = 0.0f;
        for (int i = 0; i < SEQ; ++i) {
            const float e = __expf(row[i] - m);
            row[i] = e;
            s += e;
        }
        const float inv = 1.0f / s;
        bf16* pr = P + tid * SEQ;
        for (int i = 0; i < SEQ; ++i) pr[i] = (bf16)(row[i] * inv);
    }
    wait_asynccnt0();
    __syncthreads();

    // ---- phase 3: attended = P * V, V^T tiles double-buffered in LDS ----
    v16bf aP[4];
#pragma unroll
    for (int kt = 0; kt < 4; ++kt)
        aP[kt] = load_frag16x32(P + (size_t)s0 * SEQ + kt * 32, SEQ, lane);

    bf16* Ab = att + (size_t)b * SEQ * D_EMB;

    for (int s = 0; s < 32; ++s) {
        const int  e0   = s * 32;
        const bool more = (s + 1) < 32;

        if (more) {
            const uint32_t buf = (uint32_t)((s + 1) & 1) * (32 * VLD * 2);
#pragma unroll
            for (int j = 0; j < 2; ++j) {
                const int chunk = tid + j * 256;
                const int vrow  = chunk >> 4;
                const int vch   = chunk & 15;
                async_load_b128(kt_base + buf + (uint32_t)(vrow * (VLD * 2) + vch * 16),
                                Vtb + (size_t)(e0 + 32 + vrow) * SEQ + vch * 8);
            }
        }

        const bf16* vl = vtile + (s & 1) * (32 * VLD);
#pragma unroll
        for (int c = 0; c < 2; ++c) {
            v8f cc = vzero8f();
#pragma unroll
            for (int kt = 0; kt < 4; ++kt) {
                // B[k][n] = V[kt*32+k][e0+c*16+n] = Vt[e0+c*16+n][kt*32+k]
                const v16bf bf = load_frag16x32(vl + (c * 16) * VLD + kt * 32,
                                                VLD, lane);
                cc = wmma_bf16(aP[kt], bf, cc);
            }
#pragma unroll
            for (int j = 0; j < 8; ++j)
                Ab[(size_t)(s0 + mb + j) * D_EMB + e0 + c * 16 + n] = (bf16)cc[j];
        }

        if (more) {
            wait_asynccnt0();
            __syncthreads();
        }
    }
}

// --------------------------- output projection ------------------------------
// Same blocking as qkv_gemm: 8 waves, 128x64 tile, async-staged Wo tile.
__global__ void out_gemm(const bf16* __restrict__ att,
                         const bf16* __restrict__ wo,
                         const float* __restrict__ bo,
                         float* __restrict__ out) {
    __shared__ bf16 wtile[2][64 * TLD];

    const int tid  = threadIdx.x;
    const int lane = tid & 31;
    const int w    = tid >> 5;
    const int r0   = blockIdx.x * 128 + w * 16;
    const int c0   = blockIdx.y * 64;

    const int trow = tid >> 2;
    const int tch  = tid & 3;
    const uint32_t wt_base = lds_off_u32(&wtile[0][0]);

    v8f acc[4];
#pragma unroll
    for (int t = 0; t < 4; ++t) acc[t] = vzero8f();

    const bf16* arow = att + (size_t)r0 * D_EMB;

    async_load_b128(wt_base + (uint32_t)(trow * (TLD * 2) + tch * 16),
                    wo + (size_t)(c0 + trow) * D_EMB + tch * 8);
    wait_asynccnt0();
    __syncthreads();

    v16bf a_cur = load_frag16x32(arow, D_EMB, lane);

    for (int kt = 0; kt < 32; ++kt) {
        const int  k0   = kt * 32;
        const bool more = (kt + 1) < 32;

        if (more) {
            const uint32_t buf = (uint32_t)((kt + 1) & 1) * (64 * TLD * 2);
            async_load_b128(wt_base + buf + (uint32_t)(trow * (TLD * 2) + tch * 16),
                            wo + (size_t)(c0 + trow) * D_EMB + k0 + 32 + tch * 8);
        }
        v16bf a_nxt = a_cur;
        if (more) {
            __builtin_prefetch(arow + k0 + 64, 0, 1);
            a_nxt = load_frag16x32(arow + k0 + 32, D_EMB, lane);
        }

        const bf16* wt = &wtile[kt & 1][0];
#pragma unroll
        for (int t = 0; t < 4; ++t) {
            const v16bf b = load_frag16x32(wt + t * 16 * TLD, TLD, lane);
            acc[t] = wmma_bf16(a_cur, b, acc[t]);
        }
        a_cur = a_nxt;

        if (more) {
            wait_asynccnt0();
            __syncthreads();
        }
    }

    const int n  = lane & 15;
    const int mb = (lane >> 4) * 8;
#pragma unroll
    for (int t = 0; t < 4; ++t) {
        const int   col = c0 + t * 16 + n;
        const float bb  = bo[col];
#pragma unroll
        for (int j = 0; j < 8; ++j)
            out[(size_t)(r0 + mb + j) * D_EMB + col] = acc[t][j] + bb;
    }
}

// --------------------------- launch ----------------------------------------

extern "C" void kernel_launch(void* const* d_in, const int* in_sizes, int n_in,
                              void* d_out, int out_size, void* d_ws, size_t ws_size,
                              hipStream_t stream) {
    const float* x  = (const float*)d_in[0];
    const float* Wq = (const float*)d_in[1];
    const float* bq = (const float*)d_in[2];
    const float* Wk = (const float*)d_in[3];
    const float* bk = (const float*)d_in[4];
    const float* Wv = (const float*)d_in[5];
    const float* bv = (const float*)d_in[6];
    const float* Wo = (const float*)d_in[7];
    const float* bo = (const float*)d_in[8];

    char* ws = (char*)d_ws;
    const size_t XB = (size_t)ROWS * D_EMB * sizeof(bf16);   // 32MB
    const size_t WB = (size_t)D_EMB * D_EMB * sizeof(bf16);  // 2MB
    bf16* xbf = (bf16*)(ws);
    bf16* wqb = (bf16*)(ws + XB);
    bf16* wkb = (bf16*)(ws + XB + WB);
    bf16* wvb = (bf16*)(ws + XB + 2 * WB);
    bf16* wob = (bf16*)(ws + XB + 3 * WB);
    bf16* qb  = (bf16*)(ws + XB + 4 * WB);
    bf16* kb  = (bf16*)(ws + 2 * XB + 4 * WB);
    bf16* vtb = (bf16*)(ws + 3 * XB + 4 * WB);
    bf16* attb = xbf;   // x no longer needed once Q/K/V are built

    {
        const int nx4 = (ROWS * D_EMB) / 4;
        cvt_f32_to_bf16_x4<<<(nx4 + 255) / 256, 256, 0, stream>>>(x, xbf, nx4);
        const int nw4 = (D_EMB * D_EMB) / 4;
        const int gw  = (nw4 + 255) / 256;
        cvt_f32_to_bf16_x4<<<gw, 256, 0, stream>>>(Wq, wqb, nw4);
        cvt_f32_to_bf16_x4<<<gw, 256, 0, stream>>>(Wk, wkb, nw4);
        cvt_f32_to_bf16_x4<<<gw, 256, 0, stream>>>(Wv, wvb, nw4);
        cvt_f32_to_bf16_x4<<<gw, 256, 0, stream>>>(Wo, wob, nw4);
    }

    qkv_gemm<<<dim3(ROWS / 128, D_EMB / 64, 3), 256, 0, stream>>>(
        xbf, wqb, wkb, wvb, bq, bk, bv, qb, kb, vtb);

    const size_t attn_lds = (size_t)SEQ * SLD * sizeof(float)
                          + (size_t)SEQ * SEQ * sizeof(bf16)
                          + 2 * (size_t)SEQ * TLD * sizeof(bf16);
    attn_kernel<<<NBATCH, 256, attn_lds, stream>>>(qb, kb, vtb, attb);

    out_gemm<<<dim3(ROWS / 128, D_EMB / 64), 256, 0, stream>>>(
        attb, wob, bo, (float*)d_out);
}
